// AttentionBlock_29824252903993
// MI455X (gfx1250) — compile-verified, matrix operand order checked by reading, not run on hardware
//
#include <hip/hip_runtime.h>
#include <hip/hip_bf16.h>
#include <math.h>

#define SEQ 4096
#define HID 1024
#define NUM_HEADS 16
#define HEAD_DIM 64
#define LN_EPS 1e-5f

typedef __bf16 bf16_t;
typedef __attribute__((ext_vector_type(16))) __bf16 v16bf;
typedef __attribute__((ext_vector_type(8)))  __bf16 v8bf;
typedef __attribute__((ext_vector_type(8)))  float  v8f;
typedef __attribute__((ext_vector_type(4)))  int    v4i;

// ---------------------------------------------------------------------------
// Async global->LDS copy (CDNA5 GLOBAL_LOAD_ASYNC_TO_LDS_B128, ASYNCcnt)
// Builtin signature (from hipcc diagnostic): param0 = v4i addrspace(1)* (global
// source), param1 = LDS destination, then imm offset / cpol.
// ---------------------------------------------------------------------------
#if defined(__has_builtin)
#if __has_builtin(__builtin_amdgcn_global_load_async_to_lds_b128)
#define HAVE_ASYNC_LDS 1
#endif
#if __has_builtin(__builtin_amdgcn_s_wait_asynccnt)
#define HAVE_WAIT_ASYNC 1
#endif
#endif

typedef __attribute__((address_space(1))) v4i gv4i_t;
typedef __attribute__((address_space(3))) v4i lv4i_t;

__device__ __forceinline__ void async_cp16(const void* g, void* l) {
#ifdef HAVE_ASYNC_LDS
  __builtin_amdgcn_global_load_async_to_lds_b128((gv4i_t*)g, (lv4i_t*)l, 0, 0);
#else
  *(v8bf*)l = *(const v8bf*)g;
#endif
}

#if defined(HAVE_WAIT_ASYNC)
#define WAIT_ASYNC(n) __builtin_amdgcn_s_wait_asynccnt(n)
#elif defined(HAVE_ASYNC_LDS)
#define WAIT_ASYNC(n) asm volatile("s_wait_asynccnt %0" ::"i"(n) : "memory")
#else
#define WAIT_ASYNC(n) do {} while (0)
#endif

// ---------------------------------------------------------------------------
// WMMA helpers (CDNA5 / gfx1250, wave32)
// ---------------------------------------------------------------------------
// Load a 16(row)x32(k) bf16 fragment in the WMMA A-operand layout from a
// row-major matrix (row stride = ld elements). Per ISA 7.12.2 (16-bit A 16x32):
//   lane L: row = L&15 ; k-halves kb..kb+7 and kb+16..kb+23 with kb=(L&16)?8:0
// The B operand (32x16) uses the same loader applied to B^T stored row-major.
__device__ __forceinline__ v16bf load_frag_rm(const bf16_t* __restrict__ p, int ld) {
  const int lane = threadIdx.x & 31;
  const int row  = lane & 15;
  const int kb   = (lane & 16) ? 8 : 0;
  const bf16_t* q = p + (size_t)row * ld + kb;
  v8bf lo = *(const v8bf*)(q);
  v8bf hi = *(const v8bf*)(q + 16);
  v16bf f;
#pragma unroll
  for (int i = 0; i < 8; ++i) { f[i] = lo[i]; f[8 + i] = hi[i]; }
  return f;
}

__device__ __forceinline__ v8f wmma_bf16(v16bf a, v16bf b, v8f c) {
  return __builtin_amdgcn_wmma_f32_16x16x32_bf16(
      false, a, false, b, (short)0, c, false, false);
}

// ---------------------------------------------------------------------------
// 1) f32 -> bf16 conversion
// ---------------------------------------------------------------------------
__global__ void f32_to_bf16_kernel(const float* __restrict__ in,
                                   bf16_t* __restrict__ out, int n) {
  int i = blockIdx.x * blockDim.x + threadIdx.x;
  if (i < n) out[i] = (bf16_t)in[i];
}

// ---------------------------------------------------------------------------
// 2/5) GEMM: C[M,N] = A[M,K](bf16) @ B[N,K]^T(bf16) + bias + (optional resid)
//   Block 256 threads (8 waves) computes a 128x128 tile.
//   LDS-staged (async, double buffered): A/B tiles 128x32 bf16, rows padded to
//   40 halves (80B = 20 banks -> conflict-free 16-row fragment reads, 16B
//   aligned). Each wave computes 32x64 (2x4 fragments, 8 WMMAs per k-step).
// ---------------------------------------------------------------------------
#define GEMM_PAD 40

__global__ void gemm_bf16_kernel(const bf16_t* __restrict__ A,
                                 const bf16_t* __restrict__ B,
                                 const float* __restrict__ bias,
                                 const float* __restrict__ resid,
                                 float* __restrict__ C,
                                 int M, int N, int K) {
  __shared__ bf16_t As[2][128 * GEMM_PAD];
  __shared__ bf16_t Bs[2][128 * GEMM_PAD];

  const int t    = threadIdx.x;
  const int lane = t & 31;
  const int w    = t >> 5;        // wave id 0..7
  const int wm   = w & 3;         // 4 waves along M
  const int wn   = w >> 2;        // 2 waves along N

  const int nbn = N >> 7;
  const int bm  = (int)blockIdx.x / nbn;
  const int bn  = (int)blockIdx.x % nbn;
  const int tm0 = bm * 128;
  const int tn0 = bn * 128;

  // staging map: 128 rows x 4 chunks(16B) = 512 chunks; 2 per thread per tile
  const int r0 = (t * 2) >> 2;          // row of first chunk
  const int c0 = ((t * 2) & 3) * 8;     // halves offset of first chunk
  const int r1 = (t * 2 + 1) >> 2;
  const int c1 = ((t * 2 + 1) & 3) * 8;

  auto stage = [&](int buf, int k0) {
    async_cp16(A + (size_t)(tm0 + r0) * K + k0 + c0, &As[buf][r0 * GEMM_PAD + c0]);
    async_cp16(A + (size_t)(tm0 + r1) * K + k0 + c1, &As[buf][r1 * GEMM_PAD + c1]);
    async_cp16(B + (size_t)(tn0 + r0) * K + k0 + c0, &Bs[buf][r0 * GEMM_PAD + c0]);
    async_cp16(B + (size_t)(tn0 + r1) * K + k0 + c1, &Bs[buf][r1 * GEMM_PAD + c1]);
  };

  v8f acc[2][4] = {};
  const int nk = K >> 5;
  int buf = 0;
  stage(0, 0);

  for (int ks = 0; ks < nk; ++ks) {
    if (ks + 1 < nk) {
      stage(buf ^ 1, (ks + 1) * 32);
      WAIT_ASYNC(4);   // our 4 loads for `buf` are complete (in-order per wave)
    } else {
      WAIT_ASYNC(0);
    }
    __syncthreads();   // all waves' writes to `buf` visible

    const bf16_t* as = &As[buf][(wm * 32) * GEMM_PAD];
    const bf16_t* bs = &Bs[buf][(wn * 64) * GEMM_PAD];
    v16bf a0 = load_frag_rm(as, GEMM_PAD);
    v16bf a1 = load_frag_rm(as + 16 * GEMM_PAD, GEMM_PAD);
#pragma unroll
    for (int in = 0; in < 4; ++in) {
      v16bf bfrag = load_frag_rm(bs + in * 16 * GEMM_PAD, GEMM_PAD);
      acc[0][in] = wmma_bf16(a0, bfrag, acc[0][in]);
      acc[1][in] = wmma_bf16(a1, bfrag, acc[1][in]);
    }
    buf ^= 1;
    __syncthreads();   // everyone done reading before buffer is re-staged
  }

  // C/D layout: lane -> n = lane&15 ; VGPR j -> m = j + ((lane&16)?8:0)
  const int n_in = lane & 15;
  const int mofs = (lane & 16) ? 8 : 0;
#pragma unroll
  for (int im = 0; im < 2; ++im) {
#pragma unroll
    for (int in = 0; in < 4; ++in) {
      const int col = tn0 + wn * 64 + in * 16 + n_in;
      const float bv = bias ? bias[col] : 0.0f;
#pragma unroll
      for (int j = 0; j < 8; ++j) {
        const int row = tm0 + wm * 32 + im * 16 + mofs + j;
        float v = acc[im][in][j] + bv;
        if (resid) v += resid[(size_t)row * N + col];
        C[(size_t)row * N + col] = v;
      }
    }
  }
}

// ---------------------------------------------------------------------------
// 3) RoPE (rotate-half, tiled cos/sin) + pack to bf16. Q gets 1/sqrt(d) folded
//    in. V stored transposed [HID][SEQ] so attention B-fragments are
//    contiguous.
// ---------------------------------------------------------------------------
__global__ void rope_pack_kernel(const float* __restrict__ Qf,
                                 const float* __restrict__ Kf,
                                 const float* __restrict__ Vf,
                                 bf16_t* __restrict__ Qb,
                                 bf16_t* __restrict__ Kb,
                                 bf16_t* __restrict__ Vt) {
  int e = blockIdx.x * blockDim.x + threadIdx.x;
  if (e >= SEQ * HID) return;
  const int s = e / HID;
  const int c = e % HID;
  const int d = c & (HEAD_DIM - 1);
  const int i = d & (HEAD_DIM / 2 - 1);
  const float ang = (float)s * __powf(10000.0f, -(float)i / (HEAD_DIM / 2));
  float sn, cs;
  __sincosf(ang, &sn, &cs);

  const int pe = (d < HEAD_DIM / 2) ? e + HEAD_DIM / 2 : e - HEAD_DIM / 2;
  const float q  = Qf[e];
  const float k  = Kf[e];
  const float qr = (d < HEAD_DIM / 2) ? -Qf[pe] : Qf[pe];
  const float kr = (d < HEAD_DIM / 2) ? -Kf[pe] : Kf[pe];

  Qb[e] = (bf16_t)((q * cs + qr * sn) * 0.125f);
  Kb[e] = (bf16_t)(k * cs + kr * sn);
  Vt[(size_t)c * SEQ + s] = (bf16_t)Vf[e];
}

// ---------------------------------------------------------------------------
// 4) Flash attention. One wave = one (head, 32-query tile) as two 16-query
//    sub-tiles sharing K/V fragments (2x fragment reuse). Logits computed
//    TRANSPOSED (S^T = K_tile @ Q^T): the S^T C/D fragment converts in-lane to
//    the P A-fragment for the P@V WMMA (lane L holds query L&15; key halves
//    {kb..kb+7} from tile0 accs, {kb+16..kb+23} from tile1, kb=(L&16)?8:0).
// ---------------------------------------------------------------------------
__global__ void attn_kernel(const bf16_t* __restrict__ Qb,
                            const bf16_t* __restrict__ Kb,
                            const bf16_t* __restrict__ Vt,
                            bf16_t* __restrict__ ctx) {
  const int wid  = (blockIdx.x * blockDim.x + threadIdx.x) >> 5;
  const int lane = threadIdx.x & 31;
  const int h  = wid / (SEQ / 32);
  const int q0 = (wid % (SEQ / 32)) * 32;

  const bf16_t* Qh = Qb + h * HEAD_DIM;                 // ld = HID
  const bf16_t* Kh = Kb + h * HEAD_DIM;                 // ld = HID
  const bf16_t* Vh = Vt + (size_t)h * HEAD_DIM * SEQ;   // ld = SEQ

  v16bf qf[2][2];
#pragma unroll
  for (int tq = 0; tq < 2; ++tq) {
    qf[tq][0] = load_frag_rm(Qh + (size_t)(q0 + tq * 16) * HID + 0,  HID);
    qf[tq][1] = load_frag_rm(Qh + (size_t)(q0 + tq * 16) * HID + 32, HID);
  }

  v8f acc[2][4] = {};
  float m_i[2] = {-3.0e38f, -3.0e38f};
  float l_i[2] = {0.0f, 0.0f};
  const int mofs = (lane & 16) ? 8 : 0;

  for (int kb = 0; kb < SEQ; kb += 32) {
    v16bf k00 = load_frag_rm(Kh + (size_t)kb * HID + 0,  HID);
    v16bf k01 = load_frag_rm(Kh + (size_t)kb * HID + 32, HID);
    v16bf k10 = load_frag_rm(Kh + (size_t)(kb + 16) * HID + 0,  HID);
    v16bf k11 = load_frag_rm(Kh + (size_t)(kb + 16) * HID + 32, HID);

    v16bf pf[2];
#pragma unroll
    for (int tq = 0; tq < 2; ++tq) {
      v8f s0 = {}, s1 = {};
      s0 = wmma_bf16(k00, qf[tq][0], s0);
      s0 = wmma_bf16(k01, qf[tq][1], s0);
      s1 = wmma_bf16(k10, qf[tq][0], s1);
      s1 = wmma_bf16(k11, qf[tq][1], s1);

      float mx = s0[0];
#pragma unroll
      for (int j = 0; j < 8; ++j) { mx = fmaxf(mx, s0[j]); mx = fmaxf(mx, s1[j]); }
      mx = fmaxf(mx, __shfl_xor(mx, 16, 32));
      const float m_new = fmaxf(m_i[tq], mx);
      const float alpha = __expf(m_i[tq] - m_new);

      float psum = 0.0f;
#pragma unroll
      for (int j = 0; j < 8; ++j) {
        float p0 = __expf(s0[j] - m_new); psum += p0; pf[tq][j]     = (bf16_t)p0;
        float p1 = __expf(s1[j] - m_new); psum += p1; pf[tq][8 + j] = (bf16_t)p1;
      }
      psum += __shfl_xor(psum, 16, 32);
      l_i[tq] = l_i[tq] * alpha + psum;
      m_i[tq] = m_new;

      float alpha_j[8];
#pragma unroll
      for (int j = 0; j < 8; ++j) alpha_j[j] = __shfl(alpha, mofs + j, 32);
#pragma unroll
      for (int c = 0; c < 4; ++c)
#pragma unroll
        for (int j = 0; j < 8; ++j) acc[tq][c][j] *= alpha_j[j];
    }

#pragma unroll
    for (int c = 0; c < 4; ++c) {
      v16bf vf = load_frag_rm(Vh + (size_t)(c * 16) * SEQ + kb, SEQ);
      acc[0][c] = wmma_bf16(pf[0], vf, acc[0][c]);
      acc[1][c] = wmma_bf16(pf[1], vf, acc[1][c]);
    }
  }

  const int n_in = lane & 15;
#pragma unroll
  for (int tq = 0; tq < 2; ++tq) {
    const float linv = 1.0f / l_i[tq];
    float linv_j[8];
#pragma unroll
    for (int j = 0; j < 8; ++j) linv_j[j] = __shfl(linv, mofs + j, 32);
#pragma unroll
    for (int c = 0; c < 4; ++c)
#pragma unroll
      for (int j = 0; j < 8; ++j) {
        const int row = q0 + tq * 16 + mofs + j;
        const int col = h * HEAD_DIM + c * 16 + n_in;
        ctx[(size_t)row * HID + col] = (bf16_t)(acc[tq][c][j] * linv_j[j]);
      }
  }
}

// ---------------------------------------------------------------------------
// 6) Row LayerNorm over HID
// ---------------------------------------------------------------------------
__global__ void layernorm_kernel(const float* __restrict__ res,
                                 const float* __restrict__ gamma,
                                 const float* __restrict__ beta,
                                 float* __restrict__ out) {
  const int row = blockIdx.x;
  const float* r = res + (size_t)row * HID;
  __shared__ float red[256];
  const int t = threadIdx.x;

  float sum = 0.0f;
  for (int i = t; i < HID; i += 256) sum += r[i];
  red[t] = sum; __syncthreads();
  for (int st = 128; st > 0; st >>= 1) {
    if (t < st) red[t] += red[t + st];
    __syncthreads();
  }
  const float mean = red[0] * (1.0f / HID);
  __syncthreads();

  float vs = 0.0f;
  for (int i = t; i < HID; i += 256) { float d = r[i] - mean; vs += d * d; }
  red[t] = vs; __syncthreads();
  for (int st = 128; st > 0; st >>= 1) {
    if (t < st) red[t] += red[t + st];
    __syncthreads();
  }
  const float rstd = rsqrtf(red[0] * (1.0f / HID) + LN_EPS);

  for (int i = t; i < HID; i += 256)
    out[(size_t)row * HID + i] = (r[i] - mean) * rstd * gamma[i] + beta[i];
}

// ---------------------------------------------------------------------------
// Host launch
// ---------------------------------------------------------------------------
extern "C" void kernel_launch(void* const* d_in, const int* in_sizes, int n_in,
                              void* d_out, int out_size, void* d_ws, size_t ws_size,
                              hipStream_t stream) {
  const float* x     = (const float*)d_in[0];
  const float* Wq    = (const float*)d_in[1];
  const float* bq    = (const float*)d_in[2];
  const float* Wk    = (const float*)d_in[3];
  const float* bk    = (const float*)d_in[4];
  const float* Wv    = (const float*)d_in[5];
  const float* bv    = (const float*)d_in[6];
  const float* Wo    = (const float*)d_in[7];
  const float* bo    = (const float*)d_in[8];
  const float* gamma = (const float*)d_in[9];
  const float* beta  = (const float*)d_in[10];
  float* out = (float*)d_out;

  char* ws = (char*)d_ws;
  size_t off = 0;
  bf16_t* xb  = (bf16_t*)(ws + off); off += (size_t)SEQ * HID * 2;
  bf16_t* wqb = (bf16_t*)(ws + off); off += (size_t)HID * HID * 2;
  bf16_t* wkb = (bf16_t*)(ws + off); off += (size_t)HID * HID * 2;
  bf16_t* wvb = (bf16_t*)(ws + off); off += (size_t)HID * HID * 2;
  bf16_t* wob = (bf16_t*)(ws + off); off += (size_t)HID * HID * 2;
  float*  Qf  = (float*)(ws + off);  off += (size_t)SEQ * HID * 4;
  float*  Kf  = (float*)(ws + off);  off += (size_t)SEQ * HID * 4;
  float*  Vf  = (float*)(ws + off);  off += (size_t)SEQ * HID * 4;
  bf16_t* Qb  = (bf16_t*)(ws + off); off += (size_t)SEQ * HID * 2;
  bf16_t* Kb  = (bf16_t*)(ws + off); off += (size_t)SEQ * HID * 2;
  bf16_t* Vt  = (bf16_t*)(ws + off); off += (size_t)SEQ * HID * 2;
  bf16_t* ctx = (bf16_t*)(ws + off); off += (size_t)SEQ * HID * 2;
  float*  res = Qf;  // alias: Qf dead after rope_pack

  const int nx = SEQ * HID;
  const int nw = HID * HID;

  f32_to_bf16_kernel<<<(nx + 255) / 256, 256, 0, stream>>>(x,  xb,  nx);
  f32_to_bf16_kernel<<<(nw + 255) / 256, 256, 0, stream>>>(Wq, wqb, nw);
  f32_to_bf16_kernel<<<(nw + 255) / 256, 256, 0, stream>>>(Wk, wkb, nw);
  f32_to_bf16_kernel<<<(nw + 255) / 256, 256, 0, stream>>>(Wv, wvb, nw);
  f32_to_bf16_kernel<<<(nw + 255) / 256, 256, 0, stream>>>(Wo, wob, nw);

  // 128x128 tiles: (4096/128)*(1024/128) = 256 blocks of 256 threads
  const int gemm_blocks = (SEQ / 128) * (HID / 128);
  gemm_bf16_kernel<<<gemm_blocks, 256, 0, stream>>>(xb, wqb, bq, nullptr, Qf, SEQ, HID, HID);
  gemm_bf16_kernel<<<gemm_blocks, 256, 0, stream>>>(xb, wkb, bk, nullptr, Kf, SEQ, HID, HID);
  gemm_bf16_kernel<<<gemm_blocks, 256, 0, stream>>>(xb, wvb, bv, nullptr, Vf, SEQ, HID, HID);

  rope_pack_kernel<<<(nx + 255) / 256, 256, 0, stream>>>(Qf, Kf, Vf, Qb, Kb, Vt);

  // 16 heads * 128 query tiles = 2048 waves -> 512 blocks of 128
  attn_kernel<<<NUM_HEADS * (SEQ / 32) / 4, 128, 0, stream>>>(Qb, Kb, Vt, ctx);

  gemm_bf16_kernel<<<gemm_blocks, 256, 0, stream>>>(ctx, wob, bo, x, res, SEQ, HID, HID);

  layernorm_kernel<<<SEQ, 256, 0, stream>>>(res, gamma, beta, out);

  (void)in_sizes; (void)n_in; (void)out_size; (void)ws_size;
}
